// EncoderBlock_26431228740129
// MI455X (gfx1250) — matrix operand order repaired and checked by reading, not compiled, not run on hardware
//
#include <hip/hip_runtime.h>
#include <hip/hip_bf16.h>

// ---------------------------------------------------------------------------
// CDNA5 (gfx1250) transformer encoder block.
// - all matmuls: v_wmma_f32_16x16x32_bf16 on fragment-packed bf16 operands
// - GEMM operands staged into LDS with the Tensor Data Mover (tensor_load_to_lds
//   + s_wait_tensorcnt), double buffered; ds_load_b128 fragment reads
// - flash attention with online softmax (wave32 half-group shuffles)
// ---------------------------------------------------------------------------

typedef __attribute__((ext_vector_type(16))) __bf16 v16bf;
typedef __attribute__((ext_vector_type(8)))  float  v8f;
typedef __attribute__((ext_vector_type(4)))  unsigned int u32x4;
typedef __attribute__((ext_vector_type(8)))  int i32x8;
typedef __attribute__((ext_vector_type(4)))  int i32x4;

#define D_MODEL 512
#define SEQ     2048
#define BATCH   4
#define HEADS   8
#define DK      64
#define DFF     2048
#define ROWS    (BATCH * SEQ)          // 8192
#define HEAD_ELEMS ((size_t)SEQ * DK)  // 131072 elements per (b,h)

#if defined(__has_builtin)
#if __has_builtin(__builtin_amdgcn_tensor_load_to_lds) && defined(__gfx1250__)
#define HAVE_TDM 1
#endif
#endif
#if __has_include(<hip/amd_detail/amd_gfx1250_TDM.h>)
#define CDNA5_TDM_6ARG 1
#else
#define CDNA5_TDM_6ARG 0
#endif

// --- fragment-packed offset helpers -----------------------------------------
// A fragment (16Mx32K, bf16): lane l (half = l>>4) holds row M=l&15,
//   element j -> K = 2*((j>>1)&3) + 16*(j>>3) + 8*half + (j&1)
// Packed frag-major: 512 bf16 per frag, lane*16 + j within frag.
__device__ __forceinline__ size_t packA_off(int row, int col, int Kdim) {
  size_t frag = (size_t)(row >> 4) * (size_t)(Kdim >> 5) + (size_t)(col >> 5);
  int kp = col & 31;
  int hp = (kp >> 3) & 1;
  int j  = (kp & 7) | (((kp >> 4) & 1) << 3);
  return frag * 512 + (size_t)((((row & 15) + (hp << 4)) << 4) + j);
}
// B fragment (32Kx16N, bf16): lane l holds col N=l&15, element j -> K = (j&7) + 16*(j>>3) + 8*half
__device__ __forceinline__ size_t packB_off(int k, int n, int Kdim) {
  size_t frag = (size_t)(n >> 4) * (size_t)(Kdim >> 5) + (size_t)(k >> 5);
  int kp = k & 31;
  int hp = (kp >> 3) & 1;
  int j  = (kp & 7) | (((kp >> 4) & 1) << 3);
  return frag * 512 + (size_t)((((n & 15) + (hp << 4)) << 4) + j);
}

__device__ __forceinline__ v16bf ldfrag(const __bf16* p) {
  return *(const v16bf*)p;
}

__device__ __forceinline__ v8f wmma_bf16(v16bf a, v16bf b, v8f c) {
  return __builtin_amdgcn_wmma_f32_16x16x32_bf16(false, a, false, b,
                                                 (short)0, c, false, false);
}

// 16-lane (half-wave) reductions; masks 1..8 stay within a 16-lane group.
__device__ __forceinline__ float rmax16(float v) {
  v = fmaxf(v, __shfl_xor(v, 8, 32));
  v = fmaxf(v, __shfl_xor(v, 4, 32));
  v = fmaxf(v, __shfl_xor(v, 2, 32));
  v = fmaxf(v, __shfl_xor(v, 1, 32));
  return v;
}
__device__ __forceinline__ float rsum16(float v) {
  v += __shfl_xor(v, 8, 32);
  v += __shfl_xor(v, 4, 32);
  v += __shfl_xor(v, 2, 32);
  v += __shfl_xor(v, 1, 32);
  return v;
}

#if HAVE_TDM
// ---------------------------------------------------------------------------
// TDM 2D tile load: tile_rows rows of tile_cols8*8 bytes, row stride
// row_stride8*8 bytes, global->LDS. Descriptor per CDNA5 ISA D# layout.
// data_size = 3 (8-byte units).
// ---------------------------------------------------------------------------
__device__ __forceinline__ void tdm_load_2d(unsigned int lds_addr,
                                            const void* gptr,
                                            unsigned int tile_cols8,
                                            unsigned int tile_rows,
                                            unsigned int row_stride8,
                                            unsigned int tensor_rows) {
  unsigned long long ga = (unsigned long long)(uintptr_t)gptr;
  u32x4 g0;
  g0.x = 1u;                                   // count=1, user descriptor
  g0.y = lds_addr;                             // LDS byte address
  g0.z = (unsigned int)ga;                     // global addr [31:0]
  g0.w = (unsigned int)((ga >> 32) & 0x01FFFFFFu) | (2u << 30);  // [56:32] | type=2
  i32x8 g1;
  g1[0] = (int)(3u << 16);                                        // data_size=8B
  g1[1] = (int)((row_stride8 & 0xFFFFu) << 16);                   // tensor_dim0 lo
  g1[2] = (int)((row_stride8 >> 16) | ((tensor_rows & 0xFFFFu) << 16)); // dim0 hi | dim1 lo
  g1[3] = (int)((tensor_rows >> 16) | (tile_cols8 << 16));        // dim1 hi | tile_dim0
  g1[4] = (int)(tile_rows & 0xFFFFu);                             // tile_dim1 (tile_dim2=0)
  g1[5] = (int)row_stride8;                                       // dim0_stride lo32
  g1[6] = 0;                                                      // stride hi | dim1_stride lo
  g1[7] = 0;
  i32x4 z4 = {0, 0, 0, 0};
#if CDNA5_TDM_6ARG
  i32x8 z8 = {0, 0, 0, 0, 0, 0, 0, 0};
  __builtin_amdgcn_tensor_load_to_lds(g0, g1, z4, z4, z8, 0);
#else
  __builtin_amdgcn_tensor_load_to_lds(g0, g1, z4, z4, 0);
#endif
}
#endif

// ---------------------------------------------------------------------------
// Weight pack: fp32 [K,N] row-major -> B-fragment-packed bf16
// ---------------------------------------------------------------------------
__global__ void packw_kernel(const float* __restrict__ W, __bf16* __restrict__ out,
                             int K, int N) {
  size_t idx = (size_t)blockIdx.x * blockDim.x + threadIdx.x;
  if (idx >= (size_t)K * N) return;
  int k = (int)(idx / N);
  int n = (int)(idx % N);
  out[packB_off(k, n, K)] = (__bf16)W[idx];
}

// ---------------------------------------------------------------------------
// LayerNorm (Bessel-corrected var, eps added to std, scalar gamma/beta)
// writes A-fragment-packed bf16. One wave32 per row of 512.
// ---------------------------------------------------------------------------
__global__ __launch_bounds__(256) void ln_pack_kernel(
    const float* __restrict__ x, const float* __restrict__ gp,
    const float* __restrict__ bp, __bf16* __restrict__ outP, int rows) {
  int gid  = blockIdx.x * blockDim.x + threadIdx.x;
  int row  = gid >> 5;
  int lane = gid & 31;
  if (row >= rows) return;
  const float* xr = x + (size_t)row * D_MODEL;
  float v[16];
  float s = 0.0f;
#pragma unroll
  for (int i = 0; i < 16; ++i) { v[i] = xr[lane + (i << 5)]; s += v[i]; }
#pragma unroll
  for (int m = 16; m >= 1; m >>= 1) s += __shfl_xor(s, m, 32);
  float mean = s * (1.0f / 512.0f);
  float d2 = 0.0f;
#pragma unroll
  for (int i = 0; i < 16; ++i) { float d = v[i] - mean; d2 += d * d; }
#pragma unroll
  for (int m = 16; m >= 1; m >>= 1) d2 += __shfl_xor(d2, m, 32);
  float var   = d2 * (1.0f / 511.0f);           // unbiased (torch default)
  float scale = gp[0] / (sqrtf(var) + 1e-6f);   // eps added to std
  float beta  = bp[0];
#pragma unroll
  for (int i = 0; i < 16; ++i) {
    int col = lane + (i << 5);
    outP[packA_off(row, col, D_MODEL)] = (__bf16)((v[i] - mean) * scale + beta);
  }
}

// ---------------------------------------------------------------------------
// WMMA GEMM: C[M,N] = Apack[M,K] * Bpack[K,N] + bias, with epilogues.
// Block = 256 threads = 8 waves (4 m-waves x 2 n-waves); wave tile 32x64;
// block tile 128x128. Per 32-k step: TDM stages 8 A-frags + 8 B-frags (16 KB)
// into LDS (double buffered), then each wave does 12 ds_load_b128 + 8 WMMA.
// mode 0: packed-A bf16 output (optional relu)       [W1 -> h1pack]
// mode 1: per-head packed-A Q                         [Wq]
// mode 2: per-head packed-B K^T (contraction = dk)    [Wk]
// mode 3: per-head packed-B V  (contraction = key)    [Wv]
// mode 4: f32 row-major output + residual             [Wo, W2]
// ---------------------------------------------------------------------------
__global__ __launch_bounds__(256) void gemm_wmma_kernel(
    const __bf16* __restrict__ Ap, const __bf16* __restrict__ Bp,
    const float* __restrict__ bias, const float* __restrict__ resid,
    float* __restrict__ outF, __bf16* __restrict__ outP,
    int M, int N, int K, int mode, int relu) {
  const int lane  = threadIdx.x & 31;
  const int wave  = threadIdx.x >> 5;
  const int wm    = wave >> 1;        // 0..3
  const int wn    = wave & 1;         // 0..1
  const int lh    = (lane >> 4) & 1;
  const int l16   = lane & 15;
  const int mBase = blockIdx.y * 128 + wm * 32;
  const int nBase = blockIdx.x * 128 + wn * 64;
  const int K32   = K >> 5;
  const size_t laneOff = (size_t)lane * 16;

  v8f acc[2][4] = {};

#if HAVE_TDM
  // [buf][frag 0-7 = A tiles, 8-15 = B tiles][512 bf16]
  __shared__ __bf16 stage[2][16][512];
  const int mtb = blockIdx.y * 8;  // first m-tile of block
  const int ntb = blockIdx.x * 8;  // first n-tile of block

  // prologue: stage k-step 0 into buffer 0
  if (threadIdx.x == 0) {
    tdm_load_2d((unsigned int)(uintptr_t)&stage[0][0][0],
                Ap + (size_t)mtb * K32 * 512, 128u, 8u,
                (unsigned int)K32 * 128u, (unsigned int)(M >> 4));
    tdm_load_2d((unsigned int)(uintptr_t)&stage[0][8][0],
                Bp + (size_t)ntb * K32 * 512, 128u, 8u,
                (unsigned int)K32 * 128u, (unsigned int)(N >> 4));
  }

  for (int kk = 0; kk < K32; ++kk) {
    const int buf = kk & 1;
    if (threadIdx.x == 0) {
      if (kk + 1 < K32) {
        tdm_load_2d((unsigned int)(uintptr_t)&stage[buf ^ 1][0][0],
                    Ap + ((size_t)mtb * K32 + (kk + 1)) * 512, 128u, 8u,
                    (unsigned int)K32 * 128u, (unsigned int)(M >> 4));
        tdm_load_2d((unsigned int)(uintptr_t)&stage[buf ^ 1][8][0],
                    Bp + ((size_t)ntb * K32 + (kk + 1)) * 512, 128u, 8u,
                    (unsigned int)K32 * 128u, (unsigned int)(N >> 4));
        __builtin_amdgcn_s_wait_tensorcnt(2);  // current buf complete
      } else {
        __builtin_amdgcn_s_wait_tensorcnt(0);
      }
    }
    __syncthreads();

    v16bf a0 = ldfrag(&stage[buf][wm * 2 + 0][0] + laneOff);
    v16bf a1 = ldfrag(&stage[buf][wm * 2 + 1][0] + laneOff);
    v16bf b0 = ldfrag(&stage[buf][8 + wn * 4 + 0][0] + laneOff);
    v16bf b1 = ldfrag(&stage[buf][8 + wn * 4 + 1][0] + laneOff);
    v16bf b2 = ldfrag(&stage[buf][8 + wn * 4 + 2][0] + laneOff);
    v16bf b3 = ldfrag(&stage[buf][8 + wn * 4 + 3][0] + laneOff);
    acc[0][0] = wmma_bf16(a0, b0, acc[0][0]);
    acc[0][1] = wmma_bf16(a0, b1, acc[0][1]);
    acc[0][2] = wmma_bf16(a0, b2, acc[0][2]);
    acc[0][3] = wmma_bf16(a0, b3, acc[0][3]);
    acc[1][0] = wmma_bf16(a1, b0, acc[1][0]);
    acc[1][1] = wmma_bf16(a1, b1, acc[1][1]);
    acc[1][2] = wmma_bf16(a1, b2, acc[1][2]);
    acc[1][3] = wmma_bf16(a1, b3, acc[1][3]);
    __syncthreads();  // reader done before next TDM overwrite
  }
#else
  // fallback: direct global fragment loads
  size_t aF0 = (size_t)((mBase >> 4) + 0) * K32;
  size_t aF1 = (size_t)((mBase >> 4) + 1) * K32;
  size_t bF[4];
#pragma unroll
  for (int j = 0; j < 4; ++j) bF[j] = (size_t)((nBase >> 4) + j) * K32;
  for (int kk = 0; kk < K32; ++kk) {
    v16bf a0 = ldfrag(Ap + (aF0 + kk) * 512 + laneOff);
    v16bf a1 = ldfrag(Ap + (aF1 + kk) * 512 + laneOff);
#pragma unroll
    for (int j = 0; j < 4; ++j) {
      v16bf b = ldfrag(Bp + (bF[j] + kk) * 512 + laneOff);
      acc[0][j] = wmma_bf16(a0, b, acc[0][j]);
      acc[1][j] = wmma_bf16(a1, b, acc[1][j]);
    }
  }
#endif

#pragma unroll
  for (int i = 0; i < 2; ++i) {
#pragma unroll
    for (int j = 0; j < 4; ++j) {
      int nn = nBase + j * 16 + l16;
      float bi = bias[nn];
#pragma unroll
      for (int r = 0; r < 8; ++r) {
        int mm  = mBase + i * 16 + r + (lh << 3);  // C layout: M = r + 8*half
        float v = acc[i][j][r] + bi;
        if (relu) v = fmaxf(v, 0.0f);
        if (mode == 0) {
          outP[packA_off(mm, nn, N)] = (__bf16)v;
        } else if (mode == 1) {
          int b_ = mm >> 11, s_ = mm & 2047, h_ = nn >> 6, d_ = nn & 63;
          outP[(size_t)(b_ * HEADS + h_) * HEAD_ELEMS + packA_off(s_, d_, DK)] = (__bf16)v;
        } else if (mode == 2) {
          int b_ = mm >> 11, s_ = mm & 2047, h_ = nn >> 6, d_ = nn & 63;
          outP[(size_t)(b_ * HEADS + h_) * HEAD_ELEMS + packB_off(d_, s_, DK)] = (__bf16)v;
        } else if (mode == 3) {
          int b_ = mm >> 11, s_ = mm & 2047, h_ = nn >> 6, d_ = nn & 63;
          outP[(size_t)(b_ * HEADS + h_) * HEAD_ELEMS + packB_off(s_, d_, SEQ)] = (__bf16)v;
        } else {
          size_t o = (size_t)mm * N + nn;
          outF[o] = v + resid[o];
        }
      }
    }
  }
}

// ---------------------------------------------------------------------------
// Flash attention: 4 waves/block, each wave owns one 16-query tile.
// Per 32-key block: 4 WMMA for Q*K^T (dk=64), online softmax, P repacked to
// A-fragment layout through wave-private LDS, 4 WMMA for P*V.
// Output written A-fragment-packed for the Wo GEMM.
// ---------------------------------------------------------------------------
__global__ __launch_bounds__(128) void attn_kernel(
    const __bf16* __restrict__ Qp, const __bf16* __restrict__ KTp,
    const __bf16* __restrict__ Vp, const int* __restrict__ mask,
    __bf16* __restrict__ outP) {
  __shared__ __bf16 plds[4][512];

  const int lane = threadIdx.x & 31;
  const int wave = threadIdx.x >> 5;
  const int lh   = lane >> 4;
  const int l16  = lane & 15;
  const int head = blockIdx.x >> 5;   // 32 blocks per head (2048/64 rows)
  const int qgrp = blockIdx.x & 31;
  const int bi   = head >> 3;
  const int hh   = head & 7;
  const int qtile = qgrp * 4 + wave;  // 16-row query tile index
  const size_t hb = (size_t)head * HEAD_ELEMS;
  const size_t laneOff = (size_t)lane * 16;

  v16bf aq0 = ldfrag(Qp + hb + ((size_t)qtile * 2 + 0) * 512 + laneOff);
  v16bf aq1 = ldfrag(Qp + hb + ((size_t)qtile * 2 + 1) * 512 + laneOff);

  v8f o0 = {}, o1 = {}, o2 = {}, o3 = {};
  float mr[8], lr[8];
#pragma unroll
  for (int r = 0; r < 8; ++r) { mr[r] = -3.0e38f; lr[r] = 0.0f; }

  const int* mrow = mask + bi * SEQ;

  for (int kb = 0; kb < 64; ++kb) {
    // K^T B-fragments: frag = keytile*2 + dk-chunk (4 KB contiguous per block)
    v16bf k00 = ldfrag(KTp + hb + ((size_t)(kb * 4 + 0)) * 512 + laneOff);
    v16bf k01 = ldfrag(KTp + hb + ((size_t)(kb * 4 + 1)) * 512 + laneOff);
    v16bf k10 = ldfrag(KTp + hb + ((size_t)(kb * 4 + 2)) * 512 + laneOff);
    v16bf k11 = ldfrag(KTp + hb + ((size_t)(kb * 4 + 3)) * 512 + laneOff);

    v8f s0 = {}, s1 = {};
    s0 = wmma_bf16(aq0, k00, s0);
    s0 = wmma_bf16(aq1, k01, s0);
    s1 = wmma_bf16(aq0, k10, s1);
    s1 = wmma_bf16(aq1, k11, s1);

    int key  = kb * 32 + l16;
    bool ok0 = mrow[key] != 0;
    bool ok1 = mrow[key + 16] != 0;

#pragma unroll
    for (int r = 0; r < 8; ++r) {
      float a = s0[r] * 0.125f;  if (!ok0) a = -1.0e9f;
      float c = s1[r] * 0.125f;  if (!ok1) c = -1.0e9f;
      float mx = rmax16(fmaxf(a, c));
      float mn = fmaxf(mr[r], mx);
      float al = __expf(mr[r] - mn);
      float p0 = __expf(a - mn);
      float p1 = __expf(c - mn);
      lr[r] = lr[r] * al + rsum16(p0 + p1);
      mr[r] = mn;
      o0[r] *= al; o1[r] *= al; o2[r] *= al; o3[r] *= al;
      // scatter P into A-fragment order (wave-private LDS region)
      int row = r + (lh << 3);
      {
        int col = l16;                        // keys 0..15 of block
        int hp = (col >> 3) & 1;
        int jj = col & 7;
        plds[wave][((row + (hp << 4)) << 4) + jj] = (__bf16)p0;
      }
      {
        int col = l16 + 16;                   // keys 16..31 of block
        int hp = (col >> 3) & 1;
        int jj = (col & 7) | 8;
        plds[wave][((row + (hp << 4)) << 4) + jj] = (__bf16)p1;
      }
    }

    v16bf pf = *(const v16bf*)&plds[wave][lane << 4];
    v16bf v0 = ldfrag(Vp + hb + (size_t)(0 * 64 + kb) * 512 + laneOff);
    v16bf v1 = ldfrag(Vp + hb + (size_t)(1 * 64 + kb) * 512 + laneOff);
    v16bf v2 = ldfrag(Vp + hb + (size_t)(2 * 64 + kb) * 512 + laneOff);
    v16bf v3 = ldfrag(Vp + hb + (size_t)(3 * 64 + kb) * 512 + laneOff);
    o0 = wmma_bf16(pf, v0, o0);
    o1 = wmma_bf16(pf, v1, o1);
    o2 = wmma_bf16(pf, v2, o2);
    o3 = wmma_bf16(pf, v3, o3);
  }

#pragma unroll
  for (int r = 0; r < 8; ++r) {
    float inv = 1.0f / lr[r];
    int srow = qtile * 16 + r + (lh << 3);
    int mm = bi * SEQ + srow;
    outP[packA_off(mm, hh * 64 + 0 * 16 + l16, D_MODEL)] = (__bf16)(o0[r] * inv);
    outP[packA_off(mm, hh * 64 + 1 * 16 + l16, D_MODEL)] = (__bf16)(o1[r] * inv);
    outP[packA_off(mm, hh * 64 + 2 * 16 + l16, D_MODEL)] = (__bf16)(o2[r] * inv);
    outP[packA_off(mm, hh * 64 + 3 * 16 + l16, D_MODEL)] = (__bf16)(o3[r] * inv);
  }
}

// ---------------------------------------------------------------------------
// Launch
// ---------------------------------------------------------------------------
extern "C" void kernel_launch(void* const* d_in, const int* in_sizes, int n_in,
                              void* d_out, int out_size, void* d_ws, size_t ws_size,
                              hipStream_t stream) {
  const float* x    = (const float*)d_in[0];
  const int*   msk  = (const int*)d_in[1];
  const float* Wq   = (const float*)d_in[2];
  const float* bq   = (const float*)d_in[3];
  const float* Wk   = (const float*)d_in[4];
  const float* bk   = (const float*)d_in[5];
  const float* Wv   = (const float*)d_in[6];
  const float* bv   = (const float*)d_in[7];
  const float* Wo   = (const float*)d_in[8];
  const float* bo   = (const float*)d_in[9];
  const float* W1   = (const float*)d_in[10];
  const float* b1   = (const float*)d_in[11];
  const float* W2   = (const float*)d_in[12];
  const float* b2   = (const float*)d_in[13];
  const float* g1   = (const float*)d_in[14];
  const float* be1  = (const float*)d_in[15];
  const float* g2   = (const float*)d_in[16];
  const float* be2  = (const float*)d_in[17];
  float* out = (float*)d_out;

  const size_t SZ_ACT = (size_t)ROWS * D_MODEL * 2;   // 8 MB bf16
  const size_t SZ_X2  = (size_t)ROWS * D_MODEL * 4;   // 16 MB f32
  const size_t SZ_H1  = (size_t)ROWS * DFF * 2;       // 32 MB bf16
  const size_t SZ_WS  = (size_t)D_MODEL * D_MODEL * 2;
  const size_t SZ_WF  = (size_t)D_MODEL * DFF * 2;

  char* p = (char*)d_ws;
  __bf16* xnp   = (__bf16*)p; p += SZ_ACT;
  __bf16* qp    = (__bf16*)p; p += SZ_ACT;
  __bf16* ktp   = (__bf16*)p; p += SZ_ACT;
  __bf16* vp    = (__bf16*)p; p += SZ_ACT;
  __bf16* attnp = (__bf16*)p; p += SZ_ACT;
  float*  x2    = (float*)p;  p += SZ_X2;
  __bf16* xn2p  = (__bf16*)p; p += SZ_ACT;
  __bf16* h1p   = (__bf16*)p; p += SZ_H1;
  __bf16* wqp   = (__bf16*)p; p += SZ_WS;
  __bf16* wkp   = (__bf16*)p; p += SZ_WS;
  __bf16* wvp   = (__bf16*)p; p += SZ_WS;
  __bf16* wop   = (__bf16*)p; p += SZ_WS;
  __bf16* w1p   = (__bf16*)p; p += SZ_WF;
  __bf16* w2p   = (__bf16*)p; p += SZ_WF;

  // 1) pack weights to B-fragment bf16
  packw_kernel<<<(D_MODEL * D_MODEL + 255) / 256, 256, 0, stream>>>(Wq, wqp, D_MODEL, D_MODEL);
  packw_kernel<<<(D_MODEL * D_MODEL + 255) / 256, 256, 0, stream>>>(Wk, wkp, D_MODEL, D_MODEL);
  packw_kernel<<<(D_MODEL * D_MODEL + 255) / 256, 256, 0, stream>>>(Wv, wvp, D_MODEL, D_MODEL);
  packw_kernel<<<(D_MODEL * D_MODEL + 255) / 256, 256, 0, stream>>>(Wo, wop, D_MODEL, D_MODEL);
  packw_kernel<<<(D_MODEL * DFF + 255) / 256, 256, 0, stream>>>(W1, w1p, D_MODEL, DFF);
  packw_kernel<<<(D_MODEL * DFF + 255) / 256, 256, 0, stream>>>(W2, w2p, DFF, D_MODEL);

  // 2) LN1 -> packed xn
  ln_pack_kernel<<<ROWS / 8, 256, 0, stream>>>(x, g1, be1, xnp, ROWS);

  // 3) QKV projections (epilogue packs per-head Q / K^T / V)
  dim3 g512(D_MODEL / 128, ROWS / 128);
  gemm_wmma_kernel<<<g512, 256, 0, stream>>>(xnp, wqp, bq, nullptr, nullptr, qp,
                                             ROWS, D_MODEL, D_MODEL, 1, 0);
  gemm_wmma_kernel<<<g512, 256, 0, stream>>>(xnp, wkp, bk, nullptr, nullptr, ktp,
                                             ROWS, D_MODEL, D_MODEL, 2, 0);
  gemm_wmma_kernel<<<g512, 256, 0, stream>>>(xnp, wvp, bv, nullptr, nullptr, vp,
                                             ROWS, D_MODEL, D_MODEL, 3, 0);

  // 4) flash attention -> packed-A attn output
  attn_kernel<<<BATCH * HEADS * (SEQ / 64), 128, 0, stream>>>(qp, ktp, vp, msk, attnp);

  // 5) output projection + residual -> x2 (f32)
  gemm_wmma_kernel<<<g512, 256, 0, stream>>>(attnp, wop, bo, x, x2, nullptr,
                                             ROWS, D_MODEL, D_MODEL, 4, 0);

  // 6) LN2 -> packed xn2
  ln_pack_kernel<<<ROWS / 8, 256, 0, stream>>>(x2, g2, be2, xn2p, ROWS);

  // 7) FFN: relu(xn2 @ W1 + b1) -> packed h1; then h1 @ W2 + b2 + x2 -> out
  dim3 gff(DFF / 128, ROWS / 128);
  gemm_wmma_kernel<<<gff, 256, 0, stream>>>(xn2p, w1p, b1, nullptr, nullptr, h1p,
                                            ROWS, DFF, D_MODEL, 0, 1);
  gemm_wmma_kernel<<<g512, 256, 0, stream>>>(h1p, w2p, b2, x2, out, nullptr,
                                             ROWS, D_MODEL, DFF, 4, 0);
}
